// KANLayer_50843822850530
// MI455X (gfx1250) — matrix operand order repaired and checked by reading, not compiled, not run on hardware
//
#include <hip/hip_runtime.h>
#include <math.h>

// Problem constants (match reference)
#define IN_DIM   512
#define OUT_DIM  512
#define BATCH    4096
#define NC       17
#define KPI      18                 // 17 spline coeffs + 1 bias slot per input dim
#define KTOT     (IN_DIM * KPI)    // 9216

#define GLB1 __attribute__((address_space(1)))
#define LDS3 __attribute__((address_space(3)))

typedef __bf16 v16bf __attribute__((ext_vector_type(16)));
typedef float  v8f   __attribute__((ext_vector_type(8)));
typedef int    v4i   __attribute__((ext_vector_type(4)));

union FragU { uint4 q[2]; v16bf v; };

__device__ __forceinline__ unsigned short f2bf(float f) {
    unsigned int u = __float_as_uint(f);
    u += 0x7FFFu + ((u >> 16) & 1u);     // round-to-nearest-even
    return (unsigned short)(u >> 16);
}

// ---------------------------------------------------------------------------
// Kernel 1: Wc[o][k] bf16, k = i*18 + c.  c in [0,17): w_s[i,o]*cp[i,o,c];
// c == 17: w_b[i,o].   Layout: o-major, k contiguous (matches GEMM B loads).
// ---------------------------------------------------------------------------
__global__ __launch_bounds__(256) void build_weights(const float* __restrict__ w_b,
                                                     const float* __restrict__ w_s,
                                                     const float* __restrict__ cp,
                                                     unsigned short* __restrict__ Wc) {
    const int o = blockIdx.x * 256 + threadIdx.x;  // 0..511
    const int i = blockIdx.y;                      // 0..511
    const float sv = w_s[(size_t)i * OUT_DIM + o];
    const float bv = w_b[(size_t)i * OUT_DIM + o];
    const float* c = cp + ((size_t)i * OUT_DIM + o) * NC;
    unsigned short* dst = Wc + (size_t)o * KTOT + (size_t)i * KPI;
#pragma unroll
    for (int k = 0; k < NC; ++k) dst[k] = f2bf(sv * c[k]);
    dst[NC] = f2bf(bv);
}

// ---------------------------------------------------------------------------
// Kernel 2: Ab[b][k] bf16, k = i*18 + c. c<17: cubic B-spline basis value
// (4 nonzero, analytic uniform-knot form); c==17: silu(t).
// ---------------------------------------------------------------------------
__global__ __launch_bounds__(256) void build_basis(const float* __restrict__ x,
                                                   const float* __restrict__ gp,
                                                   const float* __restrict__ bp,
                                                   unsigned short* __restrict__ Ab) {
    const int b = blockIdx.x;                 // 0..4095
    const float bound = bp[0];
    const float g0 = gp[0];
    const float invh = 1.0f / (gp[1] - gp[0]);
#pragma unroll
    for (int ii = 0; ii < 2; ++ii) {
        const int i = threadIdx.x + ii * 256; // 0..511
        float t = x[(size_t)b * IN_DIM + i];
        t = fminf(fmaxf(t, -bound), bound);
        const float sig  = 1.0f / (1.0f + __expf(-t));
        const float silu = t * sig;
        const float u  = (t - g0) * invh;
        const float fj = floorf(u);
        const int   jj = (int)fj;             // knot interval index
        const float f  = u - fj;
        const float f2v = f * f, f3v = f2v * f;
        const float omf = 1.0f - f;
        const float w0 = omf * omf * omf * (1.0f / 6.0f);
        const float w1 = (3.0f * f3v - 6.0f * f2v + 4.0f) * (1.0f / 6.0f);
        const float w2 = (-3.0f * f3v + 3.0f * f2v + 3.0f * f + 1.0f) * (1.0f / 6.0f);
        const float w3 = f3v * (1.0f / 6.0f);
        const int base = jj - 3;              // basis index of w0
        unsigned short* dst = Ab + (size_t)b * KTOT + (size_t)i * KPI;
#pragma unroll
        for (int c = 0; c < NC; ++c) {
            const int d = c - base;
            float v = 0.0f;
            v = (d == 0) ? w0 : v;
            v = (d == 1) ? w1 : v;
            v = (d == 2) ? w2 : v;
            v = (d == 3) ? w3 : v;
            dst[c] = f2bf(v);
        }
        dst[NC] = f2bf(silu);
    }
}

// ---------------------------------------------------------------------------
// Kernel 3: out[b][o] = sum_k Ab[b][k] * Wc[o][k]   (bf16 WMMA, fp32 acc)
// WG = 128 threads (4 waves, 2x2), block tile 128(b) x 128(o), K-chunk 64.
// Each wave: 64x64 = 4x4 WMMA accumulators (128 acc VGPRs).
// Double-buffered LDS fed by GLOBAL_LOAD_ASYNC_TO_LDS_B128 (ASYNCcnt).
// ---------------------------------------------------------------------------
#define LSTR   72            // padded LDS row stride in bf16 elements (64 + 8)
#define NCHUNK (KTOT / 64)   // 144

__global__ __launch_bounds__(128) void kan_gemm(const unsigned short* __restrict__ Ab,
                                                const unsigned short* __restrict__ Wc,
                                                float* __restrict__ out) {
    __shared__ unsigned short smA[2][128 * LSTR];
    __shared__ unsigned short smB[2][128 * LSTR];

    const int tid  = threadIdx.x;
    const int lane = tid & 31;
    const int wave = tid >> 5;
    const int wm   = wave & 1;        // M offset wm*64
    const int wn   = wave >> 1;       // N offset wn*64
    const int lr   = lane & 15;
    const bool lo  = lane < 16;

    const int bBase = blockIdx.y * 128;   // batch (M)
    const int oBase = blockIdx.x * 128;   // out   (N)

    // Async-stage one 128x64 chunk of A and B into LDS buffer `buf`.
    auto stage = [&](int buf, int k0) {
#pragma unroll
        for (int r = 0; r < 8; ++r) {
            const int idx = tid + r * 128;   // 0..1023
            const int m  = idx >> 3;
            const int kg = idx & 7;
            __builtin_amdgcn_global_load_async_to_lds_b128(
                (GLB1 v4i*)(Ab + (size_t)(bBase + m) * KTOT + k0 + kg * 8),
                (LDS3 v4i*)(&smA[buf][m * LSTR + kg * 8]), 0, 0);
            __builtin_amdgcn_global_load_async_to_lds_b128(
                (GLB1 v4i*)(Wc + (size_t)(oBase + m) * KTOT + k0 + kg * 8),
                (LDS3 v4i*)(&smB[buf][m * LSTR + kg * 8]), 0, 0);
        }
    };

    v8f acc[4][4];
#pragma unroll
    for (int mt = 0; mt < 4; ++mt)
#pragma unroll
        for (int nt = 0; nt < 4; ++nt)
            acc[mt][nt] = (v8f){0.f, 0.f, 0.f, 0.f, 0.f, 0.f, 0.f, 0.f};

    stage(0, 0);

    for (int chunk = 0; chunk < NCHUNK; ++chunk) {
        const int cur = chunk & 1;
        if (chunk + 1 < NCHUNK) {
            stage(cur ^ 1, (chunk + 1) * 64);
            // 16 new async ops may remain outstanding; per-wave async loads
            // complete in order, so <=16 outstanding => current buffer done.
            __builtin_amdgcn_s_wait_asynccnt(16);
        } else {
            __builtin_amdgcn_s_wait_asynccnt(0);
        }
        __syncthreads();

#pragma unroll
        for (int s = 0; s < 2; ++s) {
            // A per-lane K layout: lanes 0-15 -> K {0..7,16..23}; 16-31 -> {8..15,24..31}
            const int kbA = s * 32 + (lo ? 0 : 8);
            // B per-lane K layout: lanes 0-15 -> K {0..7,16..23}; 16-31 -> {8..15,24..31}
            const int kbB = s * 32 + (lo ? 0 : 8);

            v16bf afr[4], bfr[4];
#pragma unroll
            for (int mt = 0; mt < 4; ++mt) {
                const int m = wm * 64 + mt * 16 + lr;
                FragU u;
                u.q[0] = *(const uint4*)(&smA[cur][m * LSTR + kbA]);
                u.q[1] = *(const uint4*)(&smA[cur][m * LSTR + kbA + 16]);
                afr[mt] = u.v;
            }
#pragma unroll
            for (int nt = 0; nt < 4; ++nt) {
                const int n = wn * 64 + nt * 16 + lr;
                FragU u;
                u.q[0] = *(const uint4*)(&smB[cur][n * LSTR + kbB]);
                u.q[1] = *(const uint4*)(&smB[cur][n * LSTR + kbB + 16]);
                bfr[nt] = u.v;
            }
#pragma unroll
            for (int mt = 0; mt < 4; ++mt)
#pragma unroll
                for (int nt = 0; nt < 4; ++nt)
                    acc[mt][nt] = __builtin_amdgcn_wmma_f32_16x16x32_bf16(
                        false, afr[mt], false, bfr[nt],
                        (short)0, acc[mt][nt], false, false);
        }
        __syncthreads();   // protect buffer reused by stage issued next iteration
    }

    // D layout: VGPR r, lanes 0-15: (M=r, N=lane); lanes 16-31: (M=r+8, N=lane-16)
#pragma unroll
    for (int mt = 0; mt < 4; ++mt) {
#pragma unroll
        for (int nt = 0; nt < 4; ++nt) {
            const int o    = oBase + wn * 64 + nt * 16 + lr;
            const int brow = bBase + wm * 64 + mt * 16 + (lo ? 0 : 8);
#pragma unroll
            for (int r = 0; r < 8; ++r)
                out[(size_t)(brow + r) * OUT_DIM + o] = acc[mt][nt][r];
        }
    }
}

extern "C" void kernel_launch(void* const* d_in, const int* in_sizes, int n_in,
                              void* d_out, int out_size, void* d_ws, size_t ws_size,
                              hipStream_t stream) {
    const float* x   = (const float*)d_in[0];  // (4096, 512)
    const float* w_b = (const float*)d_in[1];  // (512, 512)
    const float* w_s = (const float*)d_in[2];  // (512, 512)
    const float* cp  = (const float*)d_in[3];  // (512, 512, 17)
    const float* gp  = (const float*)d_in[4];  // (21,)
    const float* bp  = (const float*)d_in[5];  // (1,)
    float* out = (float*)d_out;                // (4096, 512) fp32

    unsigned short* Wc = (unsigned short*)d_ws;             // 512 * 9216 bf16
    unsigned short* Ab = Wc + (size_t)OUT_DIM * KTOT;       // 4096 * 9216 bf16

    build_weights<<<dim3(2, IN_DIM), 256, 0, stream>>>(w_b, w_s, cp, Wc);
    build_basis<<<dim3(BATCH), 256, 0, stream>>>(x, gp, bp, Ab);
    kan_gemm<<<dim3(OUT_DIM / 128, BATCH / 128), 128, 0, stream>>>(Ab, Wc, out);
}
